// EquivariantMultiHeadAttention_62380105008006
// MI455X (gfx1250) — matrix operand order repaired and checked by reading, not compiled, not run on hardware
//
#include <hip/hip_runtime.h>
#include <hip/hip_bf16.h>

typedef __attribute__((ext_vector_type(16))) _Float16 v16h;
typedef __attribute__((ext_vector_type(8)))  _Float16 v8h;
typedef __attribute__((ext_vector_type(4)))  _Float16 v4h;
typedef __attribute__((ext_vector_type(8)))  float    v8f;

#define H_    128
#define NH_   8
#define HD_   16
#define RBF_  64
#define CUT_HI_ 5.0f

__device__ __forceinline__ float silu_f(float x) {
    return x / (1.0f + expf(-x));
}

// ---------------------------------------------------------------------------
// WMMA GEMM with LDS staging: C[M,Nc] = act(A[M,K] @ B[K,Nc] + bias)
// A, B row-major f32 in global.
// Block = 128 threads (4 waves). Block computes a 16(M) x 64(N) strip;
// wave w owns columns [n0 + 16w, n0 + 16w + 16).
// Per K-step (32):
//   - stage A 16x32 tile into LDS as f16, row-major  (coalesced b128 loads)
//   - stage B 32x64 tile into LDS as f16, n-major/k-contiguous
//   - fragments read as 16B-contiguous LDS vectors (ds_load_b128)
// M % 16 == 0, Nc % 64 == 0, K % 32 == 0. Grid sized exactly (no guard,
// EXEC all-ones around WMMA).
// ---------------------------------------------------------------------------
__global__ void wmma_gemm_kernel(const float* __restrict__ A,
                                 const float* __restrict__ B,
                                 const float* __restrict__ bias,
                                 float* __restrict__ C,
                                 int NB /* Nc/64 */, int K,
                                 int lda, int ldb, int ldc, int act)
{
    __shared__ _Float16 Als[16 * 32];   // [row][k]      1 KB
    __shared__ _Float16 Bls[64 * 32];   // [nLocal][k]   4 KB

    const int tid  = threadIdx.x;
    const int wave = tid >> 5;
    const int lane = tid & 31;

    const int mt = blockIdx.x / NB;
    const int nb = blockIdx.x - mt * NB;
    const int m0     = mt << 4;
    const int nBlk0  = nb << 6;            // 64-wide strip base
    const int n0     = nBlk0 + (wave << 4);

    // staging coordinates
    const int srow = tid >> 3;             // 0..15  (A row)
    const int sc4  = (tid & 7) << 2;       // 0,4,..,28 (A col group)

    // fragment coordinates (ISA 7.12.2 layouts)
    const int rowA = lane & 15;
    const int sel  = (lane >> 4) & 1;      // A: K halves {0..7,16..23}/{8..15,24..31}
    const int colL = lane & 15;            // B/C column within wave tile
    const int cw   = (wave << 4) + colL;   // column within 64-strip

    v8f acc = {};

    for (int kb = 0; kb < K; kb += 32) {
        // ---- stage A: 16 rows x 32 cols, one float4 per thread ----
        {
            const float* ap = A + (size_t)(m0 + srow) * lda + kb + sc4;
            const float4 av = *(const float4*)ap;
            v4h ah;
            ah[0] = (_Float16)av.x; ah[1] = (_Float16)av.y;
            ah[2] = (_Float16)av.z; ah[3] = (_Float16)av.w;
            *(v4h*)&Als[srow * 32 + sc4] = ah;
            if (kb + 32 < K) __builtin_prefetch(ap + 32, 0, 3);
        }
        // ---- stage B: 32 rows x 64 cols, transposed into [n][k] ----
#pragma unroll
        for (int i = 0; i < 4; ++i) {
            const int idx = i * 128 + tid;
            const int kr  = idx >> 4;            // 0..31
            const int c4  = (idx & 15) << 2;     // 0..60
            const float* bp = B + (size_t)(kb + kr) * ldb + nBlk0 + c4;
            const float4 bv4 = *(const float4*)bp;
            Bls[(c4 + 0) * 32 + kr] = (_Float16)bv4.x;
            Bls[(c4 + 1) * 32 + kr] = (_Float16)bv4.y;
            Bls[(c4 + 2) * 32 + kr] = (_Float16)bv4.z;
            Bls[(c4 + 3) * 32 + kr] = (_Float16)bv4.w;
            if (kb + 32 < K) __builtin_prefetch(bp + (size_t)32 * ldb, 0, 3);
        }
        __syncthreads();

        // ---- build fragments from LDS (16B contiguous vector reads) ----
        v16h a, b;
        {
            const v8h a0 = *(const v8h*)&Als[rowA * 32 + sel * 8];
            const v8h a1 = *(const v8h*)&Als[rowA * 32 + 16 + sel * 8];
#pragma unroll
            for (int j = 0; j < 8; ++j) { a[j] = a0[j]; a[8 + j] = a1[j]; }
        }
        {
            const v8h b0 = *(const v8h*)&Bls[cw * 32 + sel * 8];
            const v8h b1 = *(const v8h*)&Bls[cw * 32 + 16 + sel * 8];
#pragma unroll
            for (int j = 0; j < 8; ++j) { b[j] = b0[j]; b[8 + j] = b1[j]; }
        }

        acc = __builtin_amdgcn_wmma_f32_16x16x32_f16(
            /*neg_a=*/false, a, /*neg_b=*/false, b,
            /*c_mod=*/(short)0, acc, /*reuse_a=*/false, /*reuse_b=*/false);

        __syncthreads();
    }

    // C layout: VGPR r -> M = r + (lane>=16 ? 8 : 0), N = lane & 15
    const int rowOff = sel << 3;
    const int col = n0 + colL;
    const float bv = bias ? bias[col] : 0.0f;
#pragma unroll
    for (int r = 0; r < 8; ++r) {
        float val = acc[r] + bv;
        if (act == 1) val = silu_f(val);
        C[(size_t)(m0 + r + rowOff) * ldc + col] = val;
    }
}

// ---------------------------------------------------------------------------
// LayerNorm over H=128 channels, one node per block (128 threads).
// ---------------------------------------------------------------------------
__global__ void ln_x_kernel(const float* __restrict__ x,
                            const float* __restrict__ s,
                            const float* __restrict__ b,
                            float* __restrict__ xn)
{
    __shared__ float sh[128];
    __shared__ float mu_s, inv_s;
    const int node = blockIdx.x;
    const int t = threadIdx.x;

    const float v = x[(size_t)node * H_ + t];
    sh[t] = v;
    __syncthreads();
    for (int st = 64; st > 0; st >>= 1) {
        if (t < st) sh[t] += sh[t + st];
        __syncthreads();
    }
    if (t == 0) mu_s = sh[0] * (1.0f / H_);
    __syncthreads();

    const float d = v - mu_s;
    sh[t] = d * d;
    __syncthreads();
    for (int st = 64; st > 0; st >>= 1) {
        if (t < st) sh[t] += sh[t + st];
        __syncthreads();
    }
    if (t == 0) inv_s = rsqrtf(sh[0] * (1.0f / H_) + 1e-6f);
    __syncthreads();

    xn[(size_t)node * H_ + t] = d * inv_s * s[t] + b[t];
}

// ---------------------------------------------------------------------------
// Per-head LayerNorm over HD=16 channels; one (node,head) group per thread.
// ---------------------------------------------------------------------------
__global__ void head_ln_kernel(float* __restrict__ qk,
                               const float* __restrict__ s,
                               const float* __restrict__ b,
                               int total)
{
    const int g = blockIdx.x * blockDim.x + threadIdx.x;
    if (g >= total) return;
    float* p = qk + (size_t)g * HD_;
    float mu = 0.0f;
#pragma unroll
    for (int i = 0; i < HD_; ++i) mu += p[i];
    mu *= (1.0f / HD_);
    float var = 0.0f;
#pragma unroll
    for (int i = 0; i < HD_; ++i) { float d = p[i] - mu; var += d * d; }
    var *= (1.0f / HD_);
    const float inv = rsqrtf(var + 1e-6f);
#pragma unroll
    for (int i = 0; i < HD_; ++i) p[i] = (p[i] - mu) * inv * s[i] + b[i];
}

// ---------------------------------------------------------------------------
// vec_dot[n,h] = sum_a vp[n,a,h] * vp[n,a,H+h]   (vp is [N,3,384])
// ---------------------------------------------------------------------------
__global__ void vec_dot_kernel(const float* __restrict__ vp,
                               float* __restrict__ vd, int N)
{
    const int idx = blockIdx.x * blockDim.x + threadIdx.x;
    if (idx >= N * H_) return;
    const int n = idx >> 7;
    const int hh = idx & (H_ - 1);
    float sum = 0.0f;
#pragma unroll
    for (int a = 0; a < 3; ++a) {
        const float* row = vp + ((size_t)n * 3 + a) * (3 * H_);
        sum += row[hh] * row[H_ + hh];
    }
    vd[idx] = sum;
}

// ---------------------------------------------------------------------------
// Per-edge message + scatter-add. One edge per block, 128 threads (1 channel
// per thread). dk_c/dv_c are chunk-local (indexed by blockIdx.x).
// Reference splits v_j along axis=2 of [E,NH,3*HD]: offsets 0/16/32 inside
// each head's 48-channel slab.
// ---------------------------------------------------------------------------
__global__ void edge_msg_kernel(const float* __restrict__ q,
                                const float* __restrict__ k,
                                const float* __restrict__ v,
                                const float* __restrict__ vec,
                                const float* __restrict__ dk_c,
                                const float* __restrict__ dv_c,
                                const float* __restrict__ edge_weight,
                                const float* __restrict__ edge_vec,
                                const int* __restrict__ senders,
                                const int* __restrict__ receivers,
                                float* __restrict__ x_agg,
                                float* __restrict__ vec_agg,
                                int e0)
{
    const int bid = blockIdx.x;
    const int e = e0 + bid;
    const int hh = threadIdx.x;          // 0..127
    const int head = hh >> 4;
    const int d = hh & 15;

    const int s = senders[e];
    const int r = receivers[e];

    // attention logit: per-head reduction over d=16 (16-lane groups in wave32)
    float p = q[(size_t)r * H_ + hh] * k[(size_t)s * H_ + hh]
            * dk_c[(size_t)bid * H_ + hh];
    p += __shfl_xor(p, 1, 32);
    p += __shfl_xor(p, 2, 32);
    p += __shfl_xor(p, 4, 32);
    p += __shfl_xor(p, 8, 32);

    const float w = edge_weight[e];
    const float cut = (w < CUT_HI_)
        ? 0.5f * (cosf(w * (3.14159265358979323846f / CUT_HI_)) + 1.0f)
        : 0.0f;
    const float attn = silu_f(p) * cut;

    // v and dv: [*, NH, 3*HD] -> head slab of 48, sub-offsets 0/16/32
    const size_t vb  = (size_t)s  * (3 * H_) + head * (3 * HD_) + d;
    const size_t dvb = (size_t)bid * (3 * H_) + head * (3 * HD_) + d;
    const float xm = v[vb]            * dv_c[dvb]            * attn;
    const float m1 = v[vb + HD_]      * dv_c[dvb + HD_];
    const float m2 = v[vb + 2 * HD_]  * dv_c[dvb + 2 * HD_];

    atomicAdd(&x_agg[(size_t)r * H_ + hh], xm);
#pragma unroll
    for (int a = 0; a < 3; ++a) {
        const float vm = vec[((size_t)s * 3 + a) * H_ + hh] * m1
                       + m2 * edge_vec[(size_t)e * 3 + a];
        atomicAdd(&vec_agg[((size_t)r * 3 + a) * H_ + hh], vm);
    }
}

// ---------------------------------------------------------------------------
// Final: dx = vec_dot*o2 + o3 ; dvec = vec3*o1 + vec_agg
// vec3 = vp[..., 256:384]
// ---------------------------------------------------------------------------
__global__ void final_kernel(const float* __restrict__ vp,
                             const float* __restrict__ vec_dot,
                             const float* __restrict__ vec_agg,
                             const float* __restrict__ o,
                             float* __restrict__ dx,
                             float* __restrict__ dvec, int N)
{
    const int idx = blockIdx.x * blockDim.x + threadIdx.x;
    if (idx >= N * H_) return;
    const int n = idx >> 7;
    const int hh = idx & (H_ - 1);

    const float* ob = o + (size_t)n * (3 * H_);
    const float o1 = ob[hh];
    const float o2 = ob[H_ + hh];
    const float o3 = ob[2 * H_ + hh];

    dx[idx] = vec_dot[idx] * o2 + o3;
#pragma unroll
    for (int a = 0; a < 3; ++a) {
        const size_t row = (size_t)n * 3 + a;
        dvec[row * H_ + hh] =
            vp[row * (3 * H_) + 2 * H_ + hh] * o1 + vec_agg[row * H_ + hh];
    }
}

// ---------------------------------------------------------------------------
// Host side
// ---------------------------------------------------------------------------
static inline void launch_gemm(const float* A, const float* B, const float* bias,
                               float* C, int M, int Nc, int K,
                               int lda, int ldb, int ldc, int act,
                               hipStream_t stream)
{
    const int Mt = M >> 4;
    const int NB = Nc >> 6;               // 64-wide column strips
    const int blocks = Mt * NB;
    wmma_gemm_kernel<<<blocks, 128, 0, stream>>>(A, B, bias, C, NB, K,
                                                 lda, ldb, ldc, act);
}

extern "C" void kernel_launch(void* const* d_in, const int* in_sizes, int n_in,
                              void* d_out, int out_size, void* d_ws, size_t ws_size,
                              hipStream_t stream)
{
    (void)n_in; (void)out_size; (void)ws_size;

    const float* x           = (const float*)d_in[0];
    const float* vec         = (const float*)d_in[1];
    const float* edge_weight = (const float*)d_in[2];
    const float* edge_attr   = (const float*)d_in[3];
    const float* edge_vec    = (const float*)d_in[4];
    const int*   senders     = (const int*)d_in[5];
    const int*   receivers   = (const int*)d_in[6];
    const float* ln_s  = (const float*)d_in[7];
    const float* ln_b  = (const float*)d_in[8];
    const float* Wq    = (const float*)d_in[9];
    const float* bq    = (const float*)d_in[10];
    const float* Wk    = (const float*)d_in[11];
    const float* bk    = (const float*)d_in[12];
    const float* Wv    = (const float*)d_in[13];
    const float* bv    = (const float*)d_in[14];
    const float* Wvec  = (const float*)d_in[15];
    const float* Wdk   = (const float*)d_in[16];
    const float* bdk   = (const float*)d_in[17];
    const float* Wdv   = (const float*)d_in[18];
    const float* bdv   = (const float*)d_in[19];
    const float* Wo    = (const float*)d_in[20];
    const float* bo    = (const float*)d_in[21];
    const float* qln_s = (const float*)d_in[22];
    const float* qln_b = (const float*)d_in[23];
    const float* kln_s = (const float*)d_in[24];
    const float* kln_b = (const float*)d_in[25];

    const int N = in_sizes[0] / H_;   // 20000
    const int E = in_sizes[5];        // 200000
    const int EC = 32768;             // edge chunk (multiple of 16)

    // workspace carve-up (floats)
    float* ws = (float*)d_ws;
    size_t off = 0;
    float* xn      = ws + off; off += (size_t)N * H_;
    float* qbuf    = ws + off; off += (size_t)N * H_;
    float* kbuf    = ws + off; off += (size_t)N * H_;
    float* vbuf    = ws + off; off += (size_t)N * 3 * H_;
    float* vp      = ws + off; off += (size_t)N * 3 * (3 * H_);
    float* vdot    = ws + off; off += (size_t)N * H_;
    float* x_agg   = ws + off; off += (size_t)N * H_;        // contiguous with
    float* vec_agg = ws + off; off += (size_t)N * 3 * H_;    // vec_agg for memset
    float* obuf    = ws + off; off += (size_t)N * 3 * H_;
    float* dk_c    = ws + off; off += (size_t)EC * H_;
    float* dv_c    = ws + off; off += (size_t)EC * 3 * H_;

    float* out_dx   = (float*)d_out;
    float* out_dvec = out_dx + (size_t)N * H_;

    // 1) xn = LN(x)
    ln_x_kernel<<<N, 128, 0, stream>>>(x, ln_s, ln_b, xn);

    // 2) q,k,v projections (WMMA)
    launch_gemm(xn, Wq, bq, qbuf, N, H_,     H_, H_, H_,     H_,     0, stream);
    launch_gemm(xn, Wk, bk, kbuf, N, H_,     H_, H_, H_,     H_,     0, stream);
    launch_gemm(xn, Wv, bv, vbuf, N, 3 * H_, H_, H_, 3 * H_, 3 * H_, 0, stream);

    // 3) per-head LN on q, k
    {
        const int groups = N * NH_;
        const int blocks = (groups + 255) / 256;
        head_ln_kernel<<<blocks, 256, 0, stream>>>(qbuf, qln_s, qln_b, groups);
        head_ln_kernel<<<blocks, 256, 0, stream>>>(kbuf, kln_s, kln_b, groups);
    }

    // 4) vp = vec @ Wvec  (A: [3N,128])
    launch_gemm(vec, Wvec, nullptr, vp, 3 * N, 3 * H_, H_, H_, 3 * H_, 3 * H_, 0, stream);

    // 5) vec_dot
    vec_dot_kernel<<<(N * H_ + 255) / 256, 256, 0, stream>>>(vp, vdot, N);

    // 6) zero aggregation buffers (x_agg and vec_agg contiguous)
    hipMemsetAsync(x_agg, 0, (size_t)(N * H_ + N * 3 * H_) * sizeof(float), stream);

    // 7) edge path, chunked: dk/dv GEMMs (WMMA, SiLU epilogue) + message/scatter
    for (int e0 = 0; e0 < E; e0 += EC) {
        const int ec = (E - e0 < EC) ? (E - e0) : EC;  // always multiple of 16
        launch_gemm(edge_attr + (size_t)e0 * RBF_, Wdk, bdk, dk_c,
                    ec, H_,     RBF_, RBF_, H_,     H_,     1, stream);
        launch_gemm(edge_attr + (size_t)e0 * RBF_, Wdv, bdv, dv_c,
                    ec, 3 * H_, RBF_, RBF_, 3 * H_, 3 * H_, 1, stream);
        edge_msg_kernel<<<ec, 128, 0, stream>>>(qbuf, kbuf, vbuf, vec,
                                                dk_c, dv_c, edge_weight, edge_vec,
                                                senders, receivers,
                                                x_agg, vec_agg, e0);
    }

    // 8) o = x_agg @ Wo + bo
    launch_gemm(x_agg, Wo, bo, obuf, N, 3 * H_, H_, H_, 3 * H_, 3 * H_, 0, stream);

    // 9) outputs
    final_kernel<<<(N * H_ + 255) / 256, 256, 0, stream>>>(vp, vdot, vec_agg,
                                                           obuf, out_dx, out_dvec, N);
}
